// DNC_56057913147581
// MI455X (gfx1250) — matrix operand order repaired and checked by reading, not compiled, not run on hardware
//
#include <hip/hip_runtime.h>
#include <math.h>

// ---------------- problem constants ----------------
#define BB    1024      // batch
#define TT    128       // timesteps
#define DIN   256       // input dim
#define HH    128       // hidden
#define MM    128       // memory units
#define WW    128       // memory width
#define RR    4         // read heads
#define NOUT  2
#define KCAT  896       // DIN + R*W + H = 256+512+128
#define NG    512       // 4*H gates

typedef __bf16 bf16;
typedef __attribute__((ext_vector_type(16))) __bf16 v16bf;
typedef __attribute__((ext_vector_type(8)))  __bf16 v8bf;
typedef __attribute__((ext_vector_type(8)))  float  v8f;

__device__ __forceinline__ float sigm(float x) { return 1.0f / (1.0f + expf(-x)); }

__device__ __forceinline__ v16bf cat16(v8bf lo, v8bf hi) {
    return __builtin_shufflevector(lo, hi, 0,1,2,3,4,5,6,7,8,9,10,11,12,13,14,15);
}

__device__ __forceinline__ v8f wmma_bf16(v16bf a, v16bf b, v8f c) {
    // D = A(16x32 bf16) x B(32x16 bf16) + C(f32)
    return __builtin_amdgcn_wmma_f32_16x16x32_bf16(false, a, false, b, (short)0, c,
                                                   false, false);
}

// -------- one-time prep: column softmax of memory (axis=0) -> bf16 [M][W] --------
__global__ void softmax_mem_kernel(const float* __restrict__ mem, bf16* __restrict__ msm) {
    int j = threadIdx.x;             // column, 128 threads
    float mx = -1e30f;
    for (int i = 0; i < MM; ++i) mx = fmaxf(mx, mem[i * WW + j]);
    float s = 0.0f;
    for (int i = 0; i < MM; ++i) s += expf(mem[i * WW + j] - mx);
    float inv = 1.0f / s;
    for (int i = 0; i < MM; ++i)
        msm[i * WW + j] = (bf16)(expf(mem[i * WW + j] - mx) * inv);
}

// -------- one-time prep: Wcat[g][k] = [W_ih | W_hh] in bf16, [NG][KCAT] --------
__global__ void wcat_kernel(const float* __restrict__ W_ih, const float* __restrict__ W_hh,
                            bf16* __restrict__ wcat) {
    int idx = blockIdx.x * blockDim.x + threadIdx.x;
    if (idx >= NG * KCAT) return;
    int g = idx / KCAT, k = idx - g * KCAT;
    float v = (k < DIN + RR * WW) ? W_ih[g * (DIN + RR * WW) + k]
                                  : W_hh[g * HH + (k - (DIN + RR * WW))];
    wcat[idx] = (bf16)v;
}

// -------- per-step: build activation row [x_t | rv_t | h_{t-1}] in bf16 --------
// act is [B][KCAT]; cols 768..895 (h) are written by the cell kernel for t>0.
__global__ void prepare_kernel(const float* __restrict__ x, const float* __restrict__ rv0,
                               const float* __restrict__ read, bf16* __restrict__ act, int t) {
    int idx = blockIdx.x * blockDim.x + threadIdx.x;
    if (idx >= BB * KCAT) return;
    int b = idx / KCAT, k = idx - b * KCAT;
    float v;
    if (k < DIN) {
        v = x[(size_t)b * TT * DIN + (size_t)t * DIN + k];
    } else if (k < DIN + RR * WW) {
        int q = k - DIN;                 // 0..511
        if (t == 0) {
            v = rv0[q];                  // tile(read_vectors0) -> same for every b
        } else {
            int j = q >> 7, m = q & 127;
            int sb = (4 * b + j) & (BB - 1);   // reshape-mix: read[(4b+j)%B, m]
            v = read[sb * MM + m];
        }
    } else {
        if (t != 0) return;              // h cols written by cell kernel
        v = 0.0f;                        // h0 = 0
    }
    act[(size_t)b * KCAT + k] = (bf16)v;
}

// -------- generic bf16 WMMA GEMM: C[M][N] = A[M][K](lda) * W[N][K]^T (+bias) --------
// one wave computes a 16(M) x 32(N) tile; blockDim must be a multiple of 32.
__global__ void gemm_bf16_kernel(const bf16* __restrict__ A, int lda,
                                 const bf16* __restrict__ Wt,
                                 float* __restrict__ C, int M, int N, int K,
                                 const float* __restrict__ bias0,
                                 const float* __restrict__ bias1) {
    int gw   = (blockIdx.x * blockDim.x + threadIdx.x) >> 5;
    int lane = threadIdx.x & 31;
    int nT = N >> 5;                       // 32-wide N tiles
    int mt = gw / nT;
    int nt = gw - mt * nT;
    if (mt * 16 >= M) return;              // wave-uniform guard

    int lm = lane & 15;
    int lh = lane >> 4;

    const bf16* ap  = A  + (size_t)(mt * 16 + lm) * lda;
    const bf16* bp0 = Wt + (size_t)(nt * 32 + lm) * K;
    const bf16* bp1 = bp0 + (size_t)16 * K;

    v8f acc0 = {};
    v8f acc1 = {};

    for (int k = 0; k < K; k += 32) {
        __builtin_prefetch(ap  + k + 64, 0, 1);   // global_prefetch_b8
        __builtin_prefetch(bp0 + k + 64, 0, 1);
        // A fragment: lane holds A[m, k+16h..+15] (contiguous)
        v8bf a0 = *(const v8bf*)(ap + k + 16 * lh);
        v8bf a1 = *(const v8bf*)(ap + k + 16 * lh + 8);
        v16bf av = cat16(a0, a1);
        // B fragments: lane holds W[n, k+16h..+15] (contiguous)
        v8bf w00 = *(const v8bf*)(bp0 + k + 16 * lh);
        v8bf w01 = *(const v8bf*)(bp0 + k + 16 * lh + 8);
        v16bf bv0 = cat16(w00, w01);
        v8bf w10 = *(const v8bf*)(bp1 + k + 16 * lh);
        v8bf w11 = *(const v8bf*)(bp1 + k + 16 * lh + 8);
        v16bf bv1 = cat16(w10, w11);

        acc0 = wmma_bf16(av, bv0, acc0);
        acc1 = wmma_bf16(av, bv1, acc1);
    }

    int c0 = nt * 32 + lm;
    int c1 = c0 + 16;
    float bb0 = 0.0f, bb1 = 0.0f;
    if (bias0) { bb0 += bias0[c0]; bb1 += bias0[c1]; }
    if (bias1) { bb0 += bias1[c0]; bb1 += bias1[c1]; }

#pragma unroll
    for (int v = 0; v < 8; ++v) {
        int row = mt * 16 + v + 8 * lh;    // C layout: VGPR v -> M = v + 8*(lane/16)
        C[(size_t)row * N + c0] = acc0[v] + bb0;
        C[(size_t)row * N + c1] = acc1[v] + bb1;
    }
}

// -------- per-step: LSTM pointwise; writes c, h(f32), and h(bf16) into act tail --------
__global__ void cell_kernel(const float* __restrict__ gates, float* __restrict__ c,
                            float* __restrict__ hf, bf16* __restrict__ act, int t) {
    int idx = blockIdx.x * blockDim.x + threadIdx.x;
    if (idx >= BB * HH) return;
    int b = idx >> 7, u = idx & 127;
    const float* g = gates + (size_t)b * NG;
    float gi = g[u], gf = g[HH + u], gg = g[2 * HH + u], go = g[3 * HH + u];
    float cp = (t == 0) ? 0.0f : c[idx];
    float cn = sigm(gf) * cp + sigm(gi) * tanhf(gg);
    float hn = sigm(go) * tanhf(cn);
    c[idx]  = cn;
    hf[idx] = hn;
    act[(size_t)b * KCAT + (DIN + RR * WW) + u] = (bf16)hn;   // h cols for next GEMMs
}

// -------- per-step: out_acc[b][o] += [h | rv_new] . fc_w[o]  (rv permutation inline) ----
__global__ void outacc_kernel(const float* __restrict__ hf, const float* __restrict__ read,
                              const float* __restrict__ fc_w, float* __restrict__ oacc, int t) {
    int idx = blockIdx.x * blockDim.x + threadIdx.x;
    if (idx >= BB * NOUT) return;
    int b = idx >> 1, o = idx & 1;
    const float* fw = fc_w + (size_t)o * (HH + RR * MM);
    float s = 0.0f;
    const float* hb = hf + (size_t)b * HH;
    for (int u = 0; u < HH; ++u) s += hb[u] * fw[u];
#pragma unroll
    for (int j = 0; j < RR; ++j) {
        int sb = (4 * b + j) & (BB - 1);
        const float* rr  = read + (size_t)sb * MM;
        const float* fwj = fw + HH + j * MM;
        for (int m = 0; m < MM; ++m) s += rr[m] * fwj[m];
    }
    if (t == 0) oacc[idx] = s; else oacc[idx] += s;
}

// -------- finalize: mean over T, add fc_b --------
__global__ void final_kernel(const float* __restrict__ oacc, const float* __restrict__ fc_b,
                             float* __restrict__ out) {
    int idx = blockIdx.x * blockDim.x + threadIdx.x;
    if (idx >= BB * NOUT) return;
    out[idx] = oacc[idx] * (1.0f / (float)TT) + fc_b[idx & 1];
}

// ---------------- workspace layout (bytes, 256-aligned) ----------------
#define OFF_WCAT  0u                                   // [512][896] bf16 = 917504
#define OFF_MSM   917504u                              // [128][128] bf16 = 32768
#define OFF_ACT   950272u                              // [1024][896] bf16 = 1835008
#define OFF_GATES 2785280u                             // [1024][512] f32 = 2097152
#define OFF_C     4882432u                             // [1024][128] f32 = 524288
#define OFF_HF    5406720u                             // [1024][128] f32 = 524288
#define OFF_READ  5931008u                             // [1024][128] f32 = 524288
#define OFF_OACC  6455296u                             // [1024][2]   f32 = 8192

extern "C" void kernel_launch(void* const* d_in, const int* in_sizes, int n_in,
                              void* d_out, int out_size, void* d_ws, size_t ws_size,
                              hipStream_t stream) {
    (void)in_sizes; (void)n_in; (void)out_size; (void)ws_size;
    const float* x      = (const float*)d_in[0];
    const float* mem    = (const float*)d_in[1];
    const float* rv0    = (const float*)d_in[2];
    const float* W_ih   = (const float*)d_in[3];
    const float* W_hh   = (const float*)d_in[4];
    const float* b_ih   = (const float*)d_in[5];
    const float* b_hh   = (const float*)d_in[6];
    const float* fc_w   = (const float*)d_in[7];
    const float* fc_b   = (const float*)d_in[8];

    char* ws = (char*)d_ws;
    bf16*  wcat  = (bf16*)(ws + OFF_WCAT);
    bf16*  msm   = (bf16*)(ws + OFF_MSM);
    bf16*  act   = (bf16*)(ws + OFF_ACT);
    float* gates = (float*)(ws + OFF_GATES);
    float* cbuf  = (float*)(ws + OFF_C);
    float* hf    = (float*)(ws + OFF_HF);
    float* readb = (float*)(ws + OFF_READ);
    float* oacc  = (float*)(ws + OFF_OACC);

    // one-time prep
    softmax_mem_kernel<<<1, 128, 0, stream>>>(mem, msm);
    wcat_kernel<<<(NG * KCAT + 255) / 256, 256, 0, stream>>>(W_ih, W_hh, wcat);

    // gates GEMM: waves = (1024/16)*(512/32) = 1024 -> 128 blocks of 8 waves
    // read  GEMM: waves = (1024/16)*(128/32) =  256 ->  32 blocks
    for (int t = 0; t < TT; ++t) {
        prepare_kernel<<<(BB * KCAT + 255) / 256, 256, 0, stream>>>(x, rv0, readb, act, t);
        gemm_bf16_kernel<<<128, 256, 0, stream>>>(act, KCAT, wcat, gates,
                                                  BB, NG, KCAT, b_ih, b_hh);
        cell_kernel<<<(BB * HH + 255) / 256, 256, 0, stream>>>(gates, cbuf, hf, act, t);
        gemm_bf16_kernel<<<32, 256, 0, stream>>>(act + (DIN + RR * WW), KCAT, msm, readb,
                                                 BB, MM, HH, nullptr, nullptr);
        outacc_kernel<<<(BB * NOUT + 255) / 256, 256, 0, stream>>>(hf, readb, fc_w, oacc, t);
    }
    final_kernel<<<(BB * NOUT + 255) / 256, 256, 0, stream>>>(oacc, fc_b, (float*)d_out);
}
